// GNN_30262339568140
// MI455X (gfx1250) — matrix operand order repaired and checked by reading, not compiled, not run on hardware
//
#include <hip/hip_runtime.h>
#include <hip/hip_bf16.h>

typedef __attribute__((ext_vector_type(16))) _Float16 v16h;
typedef __attribute__((ext_vector_type(8)))  _Float16 v8h;
typedef __attribute__((ext_vector_type(8)))  float    v8f;

#define TPB 256

// ---------------- small utility kernels ----------------

__global__ void zero_f32_kernel(float* __restrict__ p, long n) {
    long i = (long)blockIdx.x * blockDim.x + threadIdx.x;
    if (i < n) p[i] = 0.0f;
}

__global__ void deg_kernel(const int* __restrict__ col, float* __restrict__ deg, int E) {
    int i = blockIdx.x * blockDim.x + threadIdx.x;
    if (i < E) atomicAdd(&deg[col[i]], 1.0f);
}

// in-place: deg -> dinv = rsqrt(deg + 1)   (the +1 is the self loop; always > 0)
__global__ void dinv_kernel(float* __restrict__ d, int N) {
    int i = blockIdx.x * blockDim.x + threadIdx.x;
    if (i < N) d[i] = rsqrtf(d[i] + 1.0f);
}

__global__ void norm_kernel(const int* __restrict__ row, const int* __restrict__ col,
                            const float* __restrict__ dinv, float* __restrict__ norm, int E) {
    int i = blockIdx.x * blockDim.x + threadIdx.x;
    if (i < E) norm[i] = dinv[row[i]] * dinv[col[i]];
}

__global__ void cvt_f16_kernel(const float* __restrict__ x, _Float16* __restrict__ h, long n) {
    long i = (long)blockIdx.x * blockDim.x + threadIdx.x;
    if (i < n) h[i] = (_Float16)x[i];
}

// Pack W [K=128][Nout] (row-major f32) into per-lane WMMA B-fragment layout (f16).
// Fragment for tile (kt, nt): lane L holds col n = nt*16 + (L&15),
// K = kt*32 + (L>>4)*16 + {0..15} as 8 consecutive dwords (f16 pairs).
__global__ void packW_kernel(const float* __restrict__ W, _Float16* __restrict__ P,
                             int NT, int Nout) {
    int idx = blockIdx.x * blockDim.x + threadIdx.x;
    int total = 4 * NT * 256;               // KT=4 (K=128), 32 lanes * 8 dwords per tile
    if (idx >= total) return;
    int j    = idx & 7;
    int lane = (idx >> 3) & 31;
    int t    = idx >> 8;                    // kt*NT + nt
    int nt   = t % NT;
    int kt   = t / NT;
    int n    = nt * 16 + (lane & 15);
    int k0   = kt * 32 + ((lane >> 4) * 16) + 2 * j;
    P[(size_t)idx * 2 + 0] = (_Float16)W[(size_t)k0 * Nout + n];
    P[(size_t)idx * 2 + 1] = (_Float16)W[(size_t)(k0 + 1) * Nout + n];
}

// ---------------- WMMA GEMM: hw[rows][NT*16] = A[rows][128] @ W ----------------
// One wave per 16-row stripe; NT 16-col tiles wide; K=128 in 4 WMMA steps.

template <int NT>
__global__ __launch_bounds__(TPB)
void gemm_wmma_kernel(const _Float16* __restrict__ A, const _Float16* __restrict__ Bp,
                      float* __restrict__ C, int rows) {
    const int wave = threadIdx.x >> 5;
    const int lane = threadIdx.x & 31;
    const int m0 = (blockIdx.x * 8 + wave) * 16;
    if (m0 >= rows) return;                 // wave-uniform: EXEC stays all-ones
    const int lhalf = lane >> 4;            // 0 or 1
    const int lmod  = lane & 15;

    v8f c[NT] = {};

    const _Float16* arow = A + (size_t)(m0 + lmod) * 128 + lhalf * 16;
#pragma unroll
    for (int kt = 0; kt < 4; ++kt) {
        const v8h a0 = *(const v8h*)(arow + kt * 32);        // K = kt*32 + lhalf*16 + 0..7
        const v8h a1 = *(const v8h*)(arow + kt * 32 + 8);    // K = kt*32 + lhalf*16 + 8..15
        const v16h a = __builtin_shufflevector(a0, a1,
            0, 1, 2, 3, 4, 5, 6, 7, 8, 9, 10, 11, 12, 13, 14, 15);
#pragma unroll
        for (int nt = 0; nt < NT; ++nt) {
            const v16h b = *(const v16h*)(Bp + ((size_t)(kt * NT + nt) * 32 + lane) * 16);
            c[nt] = __builtin_amdgcn_wmma_f32_16x16x32_f16(
                false, a, false, b, (short)0, c[nt], false, false);
        }
    }

    // C/D layout: VGPR j -> row m0 + (lane>>4)*8 + j, col nt*16 + (lane&15)
    const int Nout = NT * 16;
#pragma unroll
    for (int nt = 0; nt < NT; ++nt) {
#pragma unroll
        for (int j = 0; j < 8; ++j) {
            C[(size_t)(m0 + lhalf * 8 + j) * Nout + nt * 16 + lmod] = c[nt][j];
        }
    }
}

// ---------------- edge scatter: acc[col] += hw[row] * norm ----------------

__global__ void scatter_kernel(const float* __restrict__ hw, const int* __restrict__ row,
                               const int* __restrict__ col, const float* __restrict__ norm,
                               float* __restrict__ acc, long total, int shift) {
    long idx = (long)blockIdx.x * blockDim.x + threadIdx.x;
    if (idx >= total) return;
    int  f = (int)(idx & ((1L << shift) - 1));
    long e = idx >> shift;
    int r = row[e];
    int c = col[e];
    float v = hw[((size_t)r << shift) + f] * norm[e];
    atomicAdd(acc + ((size_t)c << shift) + f, v);
}

// ---------------- epilogues: + self-loop + bias (+ReLU), emit next-layer f16 ----------------

__global__ void finish_mid_kernel(const float* __restrict__ acc, const float* __restrict__ hw,
                                  const float* __restrict__ dinv, const float* __restrict__ bias,
                                  _Float16* __restrict__ hnext, long total, int shift) {
    long i = (long)blockIdx.x * blockDim.x + threadIdx.x;
    if (i >= total) return;
    int  f    = (int)(i & ((1L << shift) - 1));
    long node = i >> shift;
    float di = dinv[node];
    float v = acc[i] + di * di * hw[i] + bias[f];
    v = v > 0.0f ? v : 0.0f;                 // ReLU
    hnext[i] = (_Float16)v;
}

__global__ void finish_out_kernel(const float* __restrict__ acc, const float* __restrict__ hw,
                                  const float* __restrict__ dinv, const float* __restrict__ bias,
                                  float* __restrict__ out, long total, int shift) {
    long i = (long)blockIdx.x * blockDim.x + threadIdx.x;
    if (i >= total) return;
    int  f    = (int)(i & ((1L << shift) - 1));
    long node = i >> shift;
    float di = dinv[node];
    out[i] = acc[i] + di * di * hw[i] + bias[f];
}

// ---------------- host-side orchestration ----------------

static inline unsigned grid_for(long n) { return (unsigned)((n + TPB - 1) / TPB); }

extern "C" void kernel_launch(void* const* d_in, const int* in_sizes, int n_in,
                              void* d_out, int out_size, void* d_ws, size_t ws_size,
                              hipStream_t stream) {
    const float* x  = (const float*)d_in[0];
    const float* W1 = (const float*)d_in[1];
    const float* b1 = (const float*)d_in[2];
    const float* W2 = (const float*)d_in[3];
    const float* b2 = (const float*)d_in[4];
    const float* W3 = (const float*)d_in[5];
    const float* b3 = (const float*)d_in[6];
    const int*   ei = (const int*)d_in[7];

    const int DIN = 128, DHID = 128, DOUT = 64;
    const int N = in_sizes[0] / DIN;         // 50000
    const int E = in_sizes[7] / 2;           // 600000
    const int* row = ei;
    const int* col = ei + E;

    // carve workspace (256B-aligned slices)
    char* ws = (char*)d_ws;
    size_t off = 0;
    auto carve = [&](size_t bytes) -> void* {
        void* p = ws + off;
        off += (bytes + 255) & ~(size_t)255;
        return p;
    };
    float*    dinv = (float*)   carve((size_t)N * sizeof(float));
    float*    norm = (float*)   carve((size_t)E * sizeof(float));
    _Float16* h16  = (_Float16*)carve((size_t)N * DHID * sizeof(_Float16));
    float*    hw   = (float*)   carve((size_t)N * DHID * sizeof(float));
    float*    acc  = (float*)   carve((size_t)N * DHID * sizeof(float));
    _Float16* Wp1  = (_Float16*)carve((size_t)DIN  * DHID * sizeof(_Float16));
    _Float16* Wp2  = (_Float16*)carve((size_t)DHID * DHID * sizeof(_Float16));
    _Float16* Wp3  = (_Float16*)carve((size_t)DHID * DOUT * sizeof(_Float16));
    (void)ws_size; (void)n_in; (void)out_size;

    // --- graph normalization ---
    zero_f32_kernel<<<grid_for(N), TPB, 0, stream>>>(dinv, N);
    deg_kernel<<<grid_for(E), TPB, 0, stream>>>(col, dinv, E);
    dinv_kernel<<<grid_for(N), TPB, 0, stream>>>(dinv, N);
    norm_kernel<<<grid_for(E), TPB, 0, stream>>>(row, col, dinv, norm, E);

    // --- input conversion + weight packing ---
    cvt_f16_kernel<<<grid_for((long)N * DIN), TPB, 0, stream>>>(x, h16, (long)N * DIN);
    packW_kernel<<<grid_for(4 * 8 * 256), TPB, 0, stream>>>(W1, Wp1, 8, DHID);
    packW_kernel<<<grid_for(4 * 8 * 256), TPB, 0, stream>>>(W2, Wp2, 8, DHID);
    packW_kernel<<<grid_for(4 * 4 * 256), TPB, 0, stream>>>(W3, Wp3, 4, DOUT);

    const unsigned gemm_blocks = (unsigned)(((N + 15) / 16 + 7) / 8);

    // --- layer 1 ---
    gemm_wmma_kernel<8><<<gemm_blocks, TPB, 0, stream>>>(h16, Wp1, hw, N);
    zero_f32_kernel<<<grid_for((long)N * DHID), TPB, 0, stream>>>(acc, (long)N * DHID);
    scatter_kernel<<<grid_for((long)E * DHID), TPB, 0, stream>>>(hw, row, col, norm, acc,
                                                                 (long)E * DHID, 7);
    finish_mid_kernel<<<grid_for((long)N * DHID), TPB, 0, stream>>>(acc, hw, dinv, b1, h16,
                                                                    (long)N * DHID, 7);

    // --- layer 2 ---
    gemm_wmma_kernel<8><<<gemm_blocks, TPB, 0, stream>>>(h16, Wp2, hw, N);
    zero_f32_kernel<<<grid_for((long)N * DHID), TPB, 0, stream>>>(acc, (long)N * DHID);
    scatter_kernel<<<grid_for((long)E * DHID), TPB, 0, stream>>>(hw, row, col, norm, acc,
                                                                 (long)E * DHID, 7);
    finish_mid_kernel<<<grid_for((long)N * DHID), TPB, 0, stream>>>(acc, hw, dinv, b2, h16,
                                                                    (long)N * DHID, 7);

    // --- layer 3 (output, no ReLU, f32) ---
    gemm_wmma_kernel<4><<<gemm_blocks, TPB, 0, stream>>>(h16, Wp3, hw, N);
    zero_f32_kernel<<<grid_for((long)N * DOUT), TPB, 0, stream>>>(acc, (long)N * DOUT);
    scatter_kernel<<<grid_for((long)E * DOUT), TPB, 0, stream>>>(hw, row, col, norm, acc,
                                                                 (long)E * DOUT, 6);
    finish_out_kernel<<<grid_for((long)N * DOUT), TPB, 0, stream>>>(acc, hw, dinv, b3,
                                                                    (float*)d_out,
                                                                    (long)N * DOUT, 6);
}
